// TensorParallelQwenAttention_48155173323279
// MI455X (gfx1250) — matrix operand order repaired and checked by reading, not compiled, not run on hardware
//
#include <hip/hip_runtime.h>
#include <hip/hip_bf16.h>
#include <math.h>

#define S_LEN 2048
#define D_DIM 2048
#define NH 16
#define NKV 2
#define HDIM 128
#define QH_COLS (NH * HDIM)   // 2048
#define KV_COLS (NKV * HDIM)  // 256

typedef _Float16 v8h_t  __attribute__((ext_vector_type(8)));
typedef _Float16 v16h_t __attribute__((ext_vector_type(16)));
typedef float    v8f_t  __attribute__((ext_vector_type(8)));

union H16U { v16h_t v; v8h_t h[2]; };

__device__ __forceinline__ v16h_t ldfrag(const _Float16* p0, const _Float16* p1) {
  H16U u;
  u.h[0] = *(const v8h_t*)p0;
  u.h[1] = *(const v8h_t*)p1;
  return u.v;
}

__device__ __forceinline__ v8f_t zero8() {
  v8f_t z;
#pragma unroll
  for (int i = 0; i < 8; ++i) z[i] = 0.0f;
  return z;
}

__device__ __forceinline__ v8f_t wmma_f16(v16h_t a, v16h_t b, v8f_t c) {
  // D = A(16x32 f16) * B(32x16 f16) + C(16x16 f32)
  return __builtin_amdgcn_wmma_f32_16x16x32_f16(false, a, false, b, (short)0, c,
                                                false, false);
}

// ---------------- elementwise conversion kernels ----------------

__global__ void cvt_f16_kernel(const float* __restrict__ in,
                               _Float16* __restrict__ out, int n) {
  int i = blockIdx.x * blockDim.x + threadIdx.x;
  if (i < n) out[i] = (_Float16)in[i];
}

// W: [K,N] f32 row-major -> Wt: [N,K] f16
__global__ void transpose_f16_kernel(const float* __restrict__ W,
                                     _Float16* __restrict__ Wt, int K, int N) {
  long long i = (long long)blockIdx.x * blockDim.x + threadIdx.x;
  long long total = (long long)K * N;
  if (i < total) {
    int k = (int)(i % K);
    int n = (int)(i / K);
    Wt[i] = (_Float16)W[(long long)k * N + n];
  }
}

// x: [S, nh*HD] f32 -> out: same layout f16, RoPE applied, optional scale
__global__ void rope_f16_kernel(const float* __restrict__ x,
                                const int* __restrict__ pos,
                                _Float16* __restrict__ out, int nh, float scl) {
  int i = blockIdx.x * blockDim.x + threadIdx.x;
  int total = S_LEN * nh * (HDIM / 2);
  if (i >= total) return;
  int p = i % (HDIM / 2);
  int h = (i / (HDIM / 2)) % nh;
  int s = i / ((HDIM / 2) * nh);
  float inv = __powf(10000.0f, -(float)p / 64.0f);
  float ang = (float)pos[s] * inv;
  float sn, cs;
  __sincosf(ang, &sn, &cs);
  size_t base = (size_t)s * (nh * HDIM) + (size_t)h * HDIM + 2 * p;
  float xe = x[base], xo = x[base + 1];
  out[base]     = (_Float16)((xe * cs - xo * sn) * scl);
  out[base + 1] = (_Float16)((xe * sn + xo * cs) * scl);
}

// V32: [S, KV_COLS] f32 -> Vt: [KV_COLS, S] f16
__global__ void vtrans_kernel(const float* __restrict__ V32,
                              _Float16* __restrict__ Vt) {
  int i = blockIdx.x * blockDim.x + threadIdx.x;
  if (i >= KV_COLS * S_LEN) return;
  int s = i % S_LEN;
  int c = i / S_LEN;
  Vt[i] = (_Float16)V32[(size_t)s * KV_COLS + c];
}

// ---------------- WMMA GEMM: C[M,N] = A[M,K] * Bt[N,K]^T + bias ----------------
// 4 waves per block; each wave computes a 64x64 tile (16 WMMAs / 16 b128 loads
// per 32-wide k-chunk). Block tile = 64 x 256.

__global__ __launch_bounds__(128)
void gemm_wmma_kernel(const _Float16* __restrict__ A,
                      const _Float16* __restrict__ Bt,
                      const float* __restrict__ bias, float* __restrict__ C,
                      int M, int N, int K) {
  const int lane = threadIdx.x & 31;
  const int wave = threadIdx.x >> 5;
  const int ln = lane & 15, hi = lane >> 4;
  const int m0 = blockIdx.x * 64;
  const int n0 = blockIdx.y * 256 + wave * 64;
  if (n0 >= N) return;

  v8f_t acc[4][4];
#pragma unroll
  for (int t = 0; t < 4; ++t)
#pragma unroll
    for (int u = 0; u < 4; ++u) acc[t][u] = zero8();

  const _Float16* a[4];
  const _Float16* b[4];
#pragma unroll
  for (int t = 0; t < 4; ++t) a[t] = A + (size_t)(m0 + 16 * t + ln) * K;
#pragma unroll
  for (int u = 0; u < 4; ++u) b[u] = Bt + (size_t)(n0 + 16 * u + ln) * K;

  for (int kk = 0; kk < K; kk += 32) {
    v16h_t af[4], bf[4];
#pragma unroll
    for (int t = 0; t < 4; ++t)
      af[t] = ldfrag(a[t] + kk + 8 * hi, a[t] + kk + 8 * hi + 16);
#pragma unroll
    for (int u = 0; u < 4; ++u)
      bf[u] = ldfrag(b[u] + kk + 8 * hi, b[u] + kk + 8 * hi + 16);
    // prefetch ~512B ahead on each stream (global_prefetch_b8)
#pragma unroll
    for (int t = 0; t < 4; ++t) __builtin_prefetch(a[t] + kk + 256, 0, 3);
#pragma unroll
    for (int u = 0; u < 4; ++u) __builtin_prefetch(b[u] + kk + 256, 0, 3);
#pragma unroll
    for (int t = 0; t < 4; ++t)
#pragma unroll
      for (int u = 0; u < 4; ++u)
        acc[t][u] = wmma_f16(af[t], bf[u], acc[t][u]);
  }

#pragma unroll
  for (int u = 0; u < 4; ++u) {
    const float bv = bias[n0 + 16 * u + ln];
#pragma unroll
    for (int t = 0; t < 4; ++t) {
#pragma unroll
      for (int v = 0; v < 8; ++v) {
        const int r = m0 + 16 * t + v + 8 * hi;
        C[(size_t)r * N + n0 + 16 * u + ln] = acc[t][u][v] + bv;
      }
    }
  }
}

// ---------------- flash attention (causal, GQA) ----------------
// One wave per (head, 16-query tile). Qh:[S,QH_COLS] f16 (pre-scaled by
// 1/sqrt(HD)), Kh:[S,KV_COLS] f16, Vt:[KV_COLS,S] f16, Ao:[S,QH_COLS] f16.

__global__ __launch_bounds__(128)
void attn_kernel(const _Float16* __restrict__ Qh, const _Float16* __restrict__ Kh,
                 const _Float16* __restrict__ Vt, _Float16* __restrict__ Ao) {
  __shared__ _Float16 Pt[4][16][40];  // per-wave 16x32 P tile, padded rows
  const int lane = threadIdx.x & 31;
  const int wave = threadIdx.x >> 5;
  const int ln = lane & 15, hi = lane >> 4;
  const int tile = blockIdx.x * 4 + wave;  // 0..2047
  const int h = tile >> 7;                 // 0..15
  const int qt = tile & 127;               // 0..127
  const int q0 = qt * 16;
  const int kvh = h >> 3;                  // GQA: 8 q-heads per kv-head

  // Q tile as 4 A-fragments (HD=128 = 4 k-chunks of 32)
  v16h_t qa[4];
  const _Float16* qp = Qh + (size_t)(q0 + ln) * QH_COLS + (size_t)h * HDIM;
#pragma unroll
  for (int c = 0; c < 4; ++c)
    qa[c] = ldfrag(qp + c * 32 + 8 * hi, qp + c * 32 + 8 * hi + 16);

  v8f_t o[8];
#pragma unroll
  for (int j = 0; j < 8; ++j) o[j] = zero8();
  float mrow[8], lrow[8];
#pragma unroll
  for (int v = 0; v < 8; ++v) { mrow[v] = -1e30f; lrow[v] = 0.0f; }

  const int nkb = (q0 + 16 + 31) >> 5;  // causal: key blocks of 32
  for (int b = 0; b < nkb; ++b) {
    const int k0 = b * 32;
    v8f_t s0 = zero8(), s1 = zero8();
    const _Float16* kp0 = Kh + (size_t)(k0 + ln) * KV_COLS + (size_t)kvh * HDIM;
    const _Float16* kp1 = Kh + (size_t)(k0 + 16 + ln) * KV_COLS + (size_t)kvh * HDIM;
#pragma unroll
    for (int c = 0; c < 4; ++c) {
      v16h_t kb0 = ldfrag(kp0 + c * 32 + 8 * hi, kp0 + c * 32 + 8 * hi + 16);
      v16h_t kb1 = ldfrag(kp1 + c * 32 + 8 * hi, kp1 + c * 32 + 8 * hi + 16);
      s0 = wmma_f16(qa[c], kb0, s0);
      s1 = wmma_f16(qa[c], kb1, s1);
    }

    float alpha[8], p0[8], p1[8];
#pragma unroll
    for (int v = 0; v < 8; ++v) {
      const int row = q0 + v + 8 * hi;
      const int c0 = k0 + ln;
      float x0 = s0[v] + (c0 > row ? -1e9f : 0.0f);
      float x1 = s1[v] + (c0 + 16 > row ? -1e9f : 0.0f);
      float mt = fmaxf(x0, x1);
      mt = fmaxf(mt, __shfl_xor(mt, 1, 32));
      mt = fmaxf(mt, __shfl_xor(mt, 2, 32));
      mt = fmaxf(mt, __shfl_xor(mt, 4, 32));
      mt = fmaxf(mt, __shfl_xor(mt, 8, 32));
      float mn = fmaxf(mrow[v], mt);
      float e0 = __expf(x0 - mn);
      float e1 = __expf(x1 - mn);
      float rs = e0 + e1;
      rs += __shfl_xor(rs, 1, 32);
      rs += __shfl_xor(rs, 2, 32);
      rs += __shfl_xor(rs, 4, 32);
      rs += __shfl_xor(rs, 8, 32);
      float al = __expf(mrow[v] - mn);
      lrow[v] = lrow[v] * al + rs;
      mrow[v] = mn;
      alpha[v] = al;
      p0[v] = e0;
      p1[v] = e1;
    }
#pragma unroll
    for (int j = 0; j < 8; ++j) {
#pragma unroll
      for (int v = 0; v < 8; ++v) o[j][v] *= alpha[v];
    }

    // P tile (D-layout) -> LDS -> A-layout fragment. Same-wave DS ops are
    // in-order, no barrier needed.
#pragma unroll
    for (int v = 0; v < 8; ++v) {
      Pt[wave][v + 8 * hi][ln]      = (_Float16)p0[v];
      Pt[wave][v + 8 * hi][ln + 16] = (_Float16)p1[v];
    }
    v16h_t pa = ldfrag(&Pt[wave][ln][8 * hi], &Pt[wave][ln][8 * hi + 16]);

#pragma unroll
    for (int j = 0; j < 8; ++j) {
      const _Float16* vp =
          Vt + (size_t)(kvh * HDIM + j * 16 + ln) * S_LEN + k0 + 8 * hi;
      v16h_t vb = ldfrag(vp, vp + 16);
      o[j] = wmma_f16(pa, vb, o[j]);
    }
  }

#pragma unroll
  for (int v = 0; v < 8; ++v) {
    const float invl = 1.0f / lrow[v];
    const size_t row = q0 + v + 8 * hi;
#pragma unroll
    for (int j = 0; j < 8; ++j)
      Ao[row * QH_COLS + (size_t)h * HDIM + j * 16 + ln] =
          (_Float16)(o[j][v] * invl);
  }
}

// ---------------- host-side launcher ----------------

extern "C" void kernel_launch(void* const* d_in, const int* in_sizes, int n_in,
                              void* d_out, int out_size, void* d_ws, size_t ws_size,
                              hipStream_t stream) {
  (void)in_sizes; (void)n_in; (void)out_size; (void)ws_size;
  const float* hidden = (const float*)d_in[0];
  // d_in[1] = attention_mask (causal, regenerated in-kernel)
  const int*   pos = (const int*)d_in[2];
  const float* Wq = (const float*)d_in[3];
  const float* bq = (const float*)d_in[4];
  const float* Wk = (const float*)d_in[5];
  const float* bk = (const float*)d_in[6];
  const float* Wv = (const float*)d_in[7];
  const float* bv = (const float*)d_in[8];
  const float* Wo = (const float*)d_in[9];
  const float* bo = (const float*)d_in[10];
  float* out = (float*)d_out;

  char* ws = (char*)d_ws;
  size_t off = 0;
  auto alloc = [&](size_t bytes) -> char* {
    char* p = ws + off;
    off = (off + bytes + 255) & ~(size_t)255;
    return p;
  };

  _Float16* Ah  = (_Float16*)alloc((size_t)S_LEN * D_DIM * 2);
  _Float16* Wqt = (_Float16*)alloc((size_t)QH_COLS * D_DIM * 2);
  _Float16* Wkt = (_Float16*)alloc((size_t)KV_COLS * D_DIM * 2);
  _Float16* Wvt = (_Float16*)alloc((size_t)KV_COLS * D_DIM * 2);
  _Float16* Wot = (_Float16*)alloc((size_t)D_DIM * QH_COLS * 2);
  float*    Q32 = (float*)alloc((size_t)S_LEN * QH_COLS * 4);
  float*    K32 = (float*)alloc((size_t)S_LEN * KV_COLS * 4);
  float*    V32 = (float*)alloc((size_t)S_LEN * KV_COLS * 4);
  _Float16* Qh  = (_Float16*)alloc((size_t)S_LEN * QH_COLS * 2);
  _Float16* Khh = (_Float16*)alloc((size_t)S_LEN * KV_COLS * 2);
  _Float16* Vth = (_Float16*)alloc((size_t)KV_COLS * S_LEN * 2);
  _Float16* Ao  = (_Float16*)alloc((size_t)S_LEN * QH_COLS * 2);

  int n = S_LEN * D_DIM;
  cvt_f16_kernel<<<(n + 255) / 256, 256, 0, stream>>>(hidden, Ah, n);
  n = D_DIM * QH_COLS;
  transpose_f16_kernel<<<(n + 255) / 256, 256, 0, stream>>>(Wq, Wqt, D_DIM, QH_COLS);
  n = D_DIM * KV_COLS;
  transpose_f16_kernel<<<(n + 255) / 256, 256, 0, stream>>>(Wk, Wkt, D_DIM, KV_COLS);
  transpose_f16_kernel<<<(n + 255) / 256, 256, 0, stream>>>(Wv, Wvt, D_DIM, KV_COLS);
  n = QH_COLS * D_DIM;
  transpose_f16_kernel<<<(n + 255) / 256, 256, 0, stream>>>(Wo, Wot, QH_COLS, D_DIM);

  gemm_wmma_kernel<<<dim3(S_LEN / 64, QH_COLS / 256), 128, 0, stream>>>(
      Ah, Wqt, bq, Q32, S_LEN, QH_COLS, D_DIM);
  gemm_wmma_kernel<<<dim3(S_LEN / 64, 1), 128, 0, stream>>>(
      Ah, Wkt, bk, K32, S_LEN, KV_COLS, D_DIM);
  gemm_wmma_kernel<<<dim3(S_LEN / 64, 1), 128, 0, stream>>>(
      Ah, Wvt, bv, V32, S_LEN, KV_COLS, D_DIM);

  const float qscale = 0.08838834764831845f;  // 1/sqrt(128)
  n = S_LEN * NH * (HDIM / 2);
  rope_f16_kernel<<<(n + 255) / 256, 256, 0, stream>>>(Q32, pos, Qh, NH, qscale);
  n = S_LEN * NKV * (HDIM / 2);
  rope_f16_kernel<<<(n + 255) / 256, 256, 0, stream>>>(K32, pos, Khh, NKV, 1.0f);
  n = KV_COLS * S_LEN;
  vtrans_kernel<<<(n + 255) / 256, 256, 0, stream>>>(V32, Vth);

  attn_kernel<<<(NH * (S_LEN / 16)) / 4, 128, 0, stream>>>(Qh, Khh, Vth, Ao);

  gemm_wmma_kernel<<<dim3(S_LEN / 64, D_DIM / 256), 128, 0, stream>>>(
      Ao, Wot, bo, out, S_LEN, D_DIM, QH_COLS);
}
